// LSTM_Attn_Decoder_Layer_79190607003840
// MI455X (gfx1250) — compile-verified, matrix operand order checked by reading, not run on hardware
//
#include <hip/hip_runtime.h>
#include <hip/hip_bf16.h>
#include <math.h>

// ---------------------------------------------------------------------------
// LSTM attention decoder, CDNA5 (gfx1250) wave32 + WMMA bf16.
// B=32, S=512, H=256.
//
// Attention (score/softmax/context) does not consume the LSTM carry, so it is
// computed fully parallel across all t; the persistent single-workgroup
// decoder only runs the truly serial per-step gate GEMMs (WMMA) + elementwise,
// with operands streamed into LDS via double-buffered
// GLOBAL_LOAD_ASYNC_TO_LDS_B128.
// ---------------------------------------------------------------------------

typedef __attribute__((ext_vector_type(16))) __bf16 v16bf;
typedef __attribute__((ext_vector_type(8)))  float  v8f;
typedef int v4i_async __attribute__((vector_size(16)));

static constexpr int B  = 32;
static constexpr int S  = 512;
static constexpr int H  = 256;
static constexpr int H2 = 512;   // 2H
static constexpr int G4 = 1024;  // 4H

__device__ __forceinline__ v8f wmma_bf16(v16bf a, v16bf b, v8f c) {
  return __builtin_amdgcn_wmma_f32_16x16x32_bf16(
      false, a, false, b, (short)0, c, false, false);
}

__device__ __forceinline__ float sigf(float x) {
  return 1.0f / (1.0f + __expf(-x));
}

// ---- async LDS copy (gfx1250) with portable fallback ----------------------
#if defined(__AMDGCN__) && __has_builtin(__builtin_amdgcn_global_load_async_to_lds_b128)
#define HAVE_ASYNC 1
#define ASYNC_CP(dst, src)                                                    \
  __builtin_amdgcn_global_load_async_to_lds_b128(                             \
      (__attribute__((address_space(1))) v4i_async*)(src),                    \
      (__attribute__((address_space(3))) v4i_async*)(dst), 0, 0)
#if __has_builtin(__builtin_amdgcn_s_wait_asynccnt)
#define ASYNC_WAIT(n) __builtin_amdgcn_s_wait_asynccnt(n)
#else
#define ASYNC_WAIT(n) asm volatile("s_wait_asynccnt %0" ::"i"(n) : "memory")
#endif
#else
#define HAVE_ASYNC 0
#define ASYNC_CP(dst, src)                                                    \
  do { *(uint4*)(dst) = *(const uint4*)(src); } while (0)
#define ASYNC_WAIT(n)                                                         \
  do { } while (0)
#endif

// ---- WMMA fragment packing (ISA 7.12.2 16-bit 16x32 A tile) ---------------
__device__ __forceinline__ void packA_store(__bf16* dst, int KT, int m, int k,
                                            float v) {
  int mt = m >> 4, kt = k >> 5, kk = k & 31;
  int khalf = (kk >> 3) & 1;
  int e = (kk & 7) + ((kk >> 4) << 3);
  int lane = (m & 15) + (khalf << 4);
  dst[(size_t)(mt * KT + kt) * 512 + lane * 16 + e] = (__bf16)v;
}

__device__ __forceinline__ void packB_store(__bf16* dst, int NT, int n, int k,
                                            float v) {
  int nt = n >> 4, kt = k >> 5, kk = k & 31;
  int khalf = (kk >> 3) & 1;
  int e = (kk & 7) + ((kk >> 4) << 3);
  int lane = (n & 15) + (khalf << 4);
  dst[(size_t)(kt * NT + nt) * 512 + lane * 16 + e] = (__bf16)v;
}

// ---- packing kernels ------------------------------------------------------

__global__ void pack_enc_kernel(const float* __restrict__ enc,
                                __bf16* __restrict__ encA) {
  int idx = blockIdx.x * blockDim.x + threadIdx.x;
  if (idx >= B * S * H) return;
  packA_store(encA, H / 32, idx / H, idx % H, enc[idx]);
}

// enc[b] (S x H) as B-operand (K=s, N=h) for the batched ctx GEMM.
__global__ void pack_encB_kernel(const float* __restrict__ enc,
                                 __bf16* __restrict__ encB) {
  int idx = blockIdx.x * blockDim.x + threadIdx.x;
  if (idx >= B * S * H) return;
  int b = idx / (S * H), s = (idx / H) % S, h = idx % H;
  packB_store(encB + (size_t)b * S * H, H / 16, h, s, enc[idx]);
}

__global__ void pack_attnW_kernel(const float* __restrict__ attn_W,
                                  __bf16* __restrict__ wq,
                                  __bf16* __restrict__ wv) {
  int idx = blockIdx.x * blockDim.x + threadIdx.x;
  if (idx >= H * H) return;
  int n = idx / H, k = idx % H;
  packB_store(wq, H / 16, n, k, attn_W[n * H2 + k]);
  packB_store(wv, H / 16, n, k, attn_W[n * H2 + H + k]);
}

__global__ void pack_weight_kernel(const float* __restrict__ W,
                                   __bf16* __restrict__ dst, int N, int K) {
  int idx = blockIdx.x * blockDim.x + threadIdx.x;
  if (idx >= N * K) return;
  packB_store(dst, N / 16, idx / K, idx % K, W[idx]);
}

// inp[t] first half = enc[:, t, :]  (A-pack, K=512)
__global__ void pack_inp_enc_kernel(const float* __restrict__ enc,
                                    __bf16* __restrict__ inp_all) {
  int idx = blockIdx.x * blockDim.x + threadIdx.x;
  if (idx >= B * S * H) return;
  int b = idx / (S * H), t = (idx / H) % S, h = idx % H;
  packA_store(inp_all + (size_t)t * (B * H2), H2 / 32, b, h, enc[idx]);
}

// ---- pq / pv precompute GEMM (M=16384, N=256, K=256) ---------------------

__global__ void __launch_bounds__(256)
gemm_pqpv_kernel(const __bf16* __restrict__ encA,
                 const __bf16* __restrict__ wq, const __bf16* __restrict__ wv,
                 const float* __restrict__ attn_b,
                 float* __restrict__ pq, float* __restrict__ pv) {
  const int wid = (blockIdx.x * blockDim.x + threadIdx.x) >> 5;
  const int lane = threadIdx.x & 31;
  const int Mt = (B * S) / 16, Nt = H / 16;
  if (wid >= 2 * Mt * Nt) return;
  const int sel = wid >= Mt * Nt;
  const int tid2 = sel ? wid - Mt * Nt : wid;
  const int mt = tid2 >> 4, nt = tid2 & 15;

  const __bf16* Bw = sel ? wv : wq;
  float binit = sel ? attn_b[nt * 16 + (lane & 15)] : 0.0f;
  v8f acc;
#pragma unroll
  for (int r = 0; r < 8; ++r) acc[r] = binit;
#pragma unroll
  for (int kt = 0; kt < H / 32; ++kt) {
    v16bf a = *(const v16bf*)(encA + (size_t)(mt * (H / 32) + kt) * 512 +
                              lane * 16);
    v16bf b = *(const v16bf*)(Bw + (size_t)(kt * Nt + nt) * 512 + lane * 16);
    acc = wmma_bf16(a, b, acc);
  }
  float* out = sel ? pv : pq;
#pragma unroll
  for (int r = 0; r < 8; ++r)
    out[(size_t)(mt * 16 + r + 8 * (lane >> 4)) * H + nt * 16 + (lane & 15)] =
        acc[r];
}

// ---- attention scores for all (b,t,s'):  fully parallel -------------------

__global__ void __launch_bounds__(256)
score_kernel(const float* __restrict__ pq, const float* __restrict__ pv,
             const float* __restrict__ attn_V, const int* __restrict__ inp_len,
             float* __restrict__ scores) {
  const int b = blockIdx.y;
  const int t0 = blockIdx.x * 8;
  __shared__ float pqs[8][H];
  __shared__ float Vs[H];
  for (int i = threadIdx.x; i < 8 * H; i += 256)
    pqs[i / H][i % H] = pq[((size_t)b * S + t0 + i / H) * H + (i % H)];
  for (int i = threadIdx.x; i < H; i += 256) Vs[i] = attn_V[i];
  __syncthreads();
  const int len = inp_len[b];
#pragma unroll
  for (int half = 0; half < 2; ++half) {
    const int s = threadIdx.x + half * 256;
    const float* pvr = pv + ((size_t)b * S + s) * H;
    float acc[8] = {0, 0, 0, 0, 0, 0, 0, 0};
    for (int h = 0; h < H; ++h) {
      if ((h & 63) == 0 && h + 64 < H)
        __builtin_prefetch(pvr + h + 64, 0, 0);  // global_prefetch_b8
      float pvv = pvr[h];
#pragma unroll
      for (int tt = 0; tt < 8; ++tt)
        acc[tt] += tanhf(pqs[tt][h] + pvv) * Vs[h];
    }
    float msk = (s < len) ? 0.0f : -1000000000.0f;
#pragma unroll
    for (int tt = 0; tt < 8; ++tt)
      scores[((size_t)b * S + t0 + tt) * S + s] = acc[tt] + msk;
  }
}

// ---- softmax over s' (wave per (b,t) row) + bf16 A-pack -------------------

__global__ void __launch_bounds__(1024)
softmax_pack_kernel(const float* __restrict__ scores,
                    __bf16* __restrict__ a_pack) {
  const int row = blockIdx.x * 32 + (threadIdx.x >> 5);  // b*S + t
  const int lane = threadIdx.x & 31;
  const int b = row / S, t = row % S;
  const float* sr = scores + (size_t)row * S;
  float vals[16];
  float mx = -INFINITY;
#pragma unroll
  for (int j = 0; j < 16; ++j) {
    vals[j] = sr[lane * 16 + j];
    mx = fmaxf(mx, vals[j]);
  }
  for (int off = 16; off; off >>= 1) mx = fmaxf(mx, __shfl_xor(mx, off, 32));
  float sum = 0.0f;
#pragma unroll
  for (int j = 0; j < 16; ++j) {
    vals[j] = __expf(vals[j] - mx);
    sum += vals[j];
  }
  for (int off = 16; off; off >>= 1) sum += __shfl_xor(sum, off, 32);
  const float inv = 1.0f / sum;
  __bf16* dst = a_pack + (size_t)b * S * S;
#pragma unroll
  for (int j = 0; j < 16; ++j)
    packA_store(dst, S / 32, t, lane * 16 + j, vals[j] * inv);
}

// ---- batched context GEMM: ctx[b] = a[b](SxS) @ enc[b](SxH), bf16 WMMA ----

__global__ void __launch_bounds__(256)
ctx_gemm_kernel(const __bf16* __restrict__ a_pack,
                const __bf16* __restrict__ encB,
                __bf16* __restrict__ inp_all, __bf16* __restrict__ x2ctx_all) {
  const int wid = (blockIdx.x * blockDim.x + threadIdx.x) >> 5;
  const int lane = threadIdx.x & 31;
  if (wid >= B * (S / 16) * (H / 16)) return;   // 32*32*16 = 16384
  const int b = wid >> 9;
  const int mt = (wid >> 4) & 31;
  const int nt = wid & 15;
  const __bf16* A = a_pack + (size_t)b * S * S;
  const __bf16* Bp = encB + (size_t)b * S * H;
  v8f acc;
#pragma unroll
  for (int r = 0; r < 8; ++r) acc[r] = 0.0f;
#pragma unroll
  for (int kt = 0; kt < S / 32; ++kt) {
    v16bf a = *(const v16bf*)(A + (size_t)(mt * (S / 32) + kt) * 512 +
                              lane * 16);
    v16bf bb = *(const v16bf*)(Bp + (size_t)(kt * (H / 16) + nt) * 512 +
                               lane * 16);
    acc = wmma_bf16(a, bb, acc);
  }
#pragma unroll
  for (int r = 0; r < 8; ++r) {
    int t = mt * 16 + r + 8 * (lane >> 4);
    int h = nt * 16 + (lane & 15);
    packA_store(inp_all + (size_t)t * (B * H2), H2 / 32, b, H + h, acc[r]);
    packA_store(x2ctx_all + (size_t)t * (B * H), H / 32, b, h, acc[r]);
  }
}

// ---- persistent recurrent decoder (1 block, 1024 threads = 32 waves) ------

__device__ __forceinline__ void issue_copies(int t, __bf16* inpDst,
                                             __bf16* x2Dst,
                                             const __bf16* __restrict__ inp_all,
                                             const __bf16* __restrict__ x2ctx_all,
                                             int tid) {
  const __bf16* s0 = inp_all + (size_t)t * (B * H2);
  ASYNC_CP(&inpDst[tid * 8], &s0[tid * 8]);
  ASYNC_CP(&inpDst[(tid + 1024) * 8], &s0[(tid + 1024) * 8]);
  // x2 ctx half: source tiles (mt*8 + kt) -> dest tiles (mt*16 + 8 + kt)
  const int tt = tid >> 6, within = tid & 63;
  const int dtile = (tt >> 3) * 16 + 8 + (tt & 7);
  ASYNC_CP(&x2Dst[dtile * 512 + within * 8],
           &x2ctx_all[(size_t)t * (B * H) + tid * 8]);
}

__global__ void __launch_bounds__(1024)
decoder_kernel(const __bf16* __restrict__ inp_all,
               const __bf16* __restrict__ x2ctx_all,
               const __bf16* __restrict__ w1ih, const __bf16* __restrict__ w1hh,
               const __bf16* __restrict__ w2ih, const __bf16* __restrict__ w2hh,
               const float* __restrict__ l1_bih, const float* __restrict__ l1_bhh,
               const float* __restrict__ l2_bih, const float* __restrict__ l2_bhh,
               const int* __restrict__ inp_len,
               const float* __restrict__ bhs, const float* __restrict__ bcs,
               float* __restrict__ out) {
  __shared__ __bf16 inpL[2][B * H2];   // 2 x 32KB
  __shared__ __bf16 x2L[2][B * H2];    // 2 x 32KB
  __shared__ __bf16 c1p[B * H];        // 16KB
  __shared__ __bf16 h2p[B * H];        // 16KB

  const int tid = threadIdx.x, wave = tid >> 5, lane = tid & 31;
  const int mtw = wave >> 4, ht = wave & 15;
  const int hcol = ht * 16 + (lane & 15);

  // carry state in registers: fixed (m,hcol) ownership for all 512 steps
  float hs1[8], cs2[8];
#pragma unroll
  for (int r = 0; r < 8; ++r) {
    const int m = mtw * 16 + r + 8 * (lane >> 4);
    hs1[r] = bhs[m * H + hcol];
    cs2[r] = 0.0f;
    packA_store(c1p, H / 32, m, hcol, bcs[m * H + hcol]);
    packA_store(h2p, H / 32, m, hcol, 0.0f);
  }
  issue_copies(0, inpL[0], x2L[0], inp_all, x2ctx_all, tid);
  __syncthreads();

  for (int t = 0; t < S; ++t) {
    const int buf = t & 1;
    if (t + 1 < S) {
      issue_copies(t + 1, inpL[buf ^ 1], x2L[buf ^ 1], inp_all, x2ctx_all, tid);
      ASYNC_WAIT(3);   // retire previous set; current prefetch stays in flight
    } else {
      ASYNC_WAIT(0);
    }
    __syncthreads();

    // ---- LSTM1 gate GEMM (note reference swaps h/c: A2 operand is c1) ----
    v8f acc[4];
#pragma unroll
    for (int g = 0; g < 4; ++g) {
      const float bb = l1_bih[g * H + hcol] + l1_bhh[g * H + hcol];
#pragma unroll
      for (int r = 0; r < 8; ++r) acc[g][r] = bb;
    }
#pragma unroll
    for (int kt = 0; kt < H2 / 32; ++kt) {
      v16bf a = *(const v16bf*)&inpL[buf][(mtw * (H2 / 32) + kt) * 512 +
                                          lane * 16];
#pragma unroll
      for (int g = 0; g < 4; ++g) {
        v16bf b = *(const v16bf*)(w1ih +
                                  (size_t)(kt * 64 + g * 16 + ht) * 512 +
                                  lane * 16);
        acc[g] = wmma_bf16(a, b, acc[g]);
      }
    }
#pragma unroll
    for (int kt = 0; kt < H / 32; ++kt) {
      v16bf a = *(const v16bf*)&c1p[(mtw * (H / 32) + kt) * 512 + lane * 16];
#pragma unroll
      for (int g = 0; g < 4; ++g) {
        v16bf b = *(const v16bf*)(w1hh +
                                  (size_t)(kt * 64 + g * 16 + ht) * 512 +
                                  lane * 16);
        acc[g] = wmma_bf16(a, b, acc[g]);
      }
    }
    __syncthreads();   // all c1p reads done before overwrite

#pragma unroll
    for (int r = 0; r < 8; ++r) {
      const int m = mtw * 16 + r + 8 * (lane >> 4);
      const float iv = sigf(acc[0][r]);
      const float fv = sigf(acc[1][r]);
      const float gv = tanhf(acc[2][r]);
      const float ov = sigf(acc[3][r]);
      const float cn = fv * hs1[r] + iv * gv;   // reference's swapped carry
      const float hn = ov * tanhf(cn);
      hs1[r] = hn;
      packA_store(c1p, H / 32, m, hcol, cn);            // next-step A operand
      packA_store(&x2L[buf][0], H2 / 32, m, hcol, hn);  // x2 dec_h half
    }
    __syncthreads();

    // ---- LSTM2 gate GEMM --------------------------------------------------
#pragma unroll
    for (int g = 0; g < 4; ++g) {
      const float bb = l2_bih[g * H + hcol] + l2_bhh[g * H + hcol];
#pragma unroll
      for (int r = 0; r < 8; ++r) acc[g][r] = bb;
    }
#pragma unroll
    for (int kt = 0; kt < H2 / 32; ++kt) {
      v16bf a = *(const v16bf*)&x2L[buf][(mtw * (H2 / 32) + kt) * 512 +
                                         lane * 16];
#pragma unroll
      for (int g = 0; g < 4; ++g) {
        v16bf b = *(const v16bf*)(w2ih +
                                  (size_t)(kt * 64 + g * 16 + ht) * 512 +
                                  lane * 16);
        acc[g] = wmma_bf16(a, b, acc[g]);
      }
    }
#pragma unroll
    for (int kt = 0; kt < H / 32; ++kt) {
      v16bf a = *(const v16bf*)&h2p[(mtw * (H / 32) + kt) * 512 + lane * 16];
#pragma unroll
      for (int g = 0; g < 4; ++g) {
        v16bf b = *(const v16bf*)(w2hh +
                                  (size_t)(kt * 64 + g * 16 + ht) * 512 +
                                  lane * 16);
        acc[g] = wmma_bf16(a, b, acc[g]);
      }
    }
    __syncthreads();   // all h2p reads done before overwrite

#pragma unroll
    for (int r = 0; r < 8; ++r) {
      const int m = mtw * 16 + r + 8 * (lane >> 4);
      const float iv = sigf(acc[0][r]);
      const float fv = sigf(acc[1][r]);
      const float gv = tanhf(acc[2][r]);
      const float ov = sigf(acc[3][r]);
      const float cn = fv * cs2[r] + iv * gv;
      const float hn = ov * tanhf(cn);
      const bool valid = t < inp_len[m];
      if (valid) {
        cs2[r] = cn;
        packA_store(h2p, H / 32, m, hcol, hn);   // h2 carry (A operand)
      }
      out[(size_t)m * S * H + (size_t)t * H + hcol] = valid ? hn : 0.0f;
    }
    __syncthreads();
  }
}

// ---------------------------------------------------------------------------

extern "C" void kernel_launch(void* const* d_in, const int* in_sizes, int n_in,
                              void* d_out, int out_size, void* d_ws,
                              size_t ws_size, hipStream_t stream) {
  (void)in_sizes; (void)n_in; (void)out_size; (void)ws_size;

  const float* enc    = (const float*)d_in[0];
  const float* bhs    = (const float*)d_in[1];
  const float* bcs    = (const float*)d_in[2];
  const float* attn_W = (const float*)d_in[3];
  const float* attn_b = (const float*)d_in[4];
  const float* attn_V = (const float*)d_in[5];
  const float* l1_Wih = (const float*)d_in[6];
  const float* l1_Whh = (const float*)d_in[7];
  const float* l1_bih = (const float*)d_in[8];
  const float* l1_bhh = (const float*)d_in[9];
  const float* l2_Wih = (const float*)d_in[10];
  const float* l2_Whh = (const float*)d_in[11];
  const float* l2_bih = (const float*)d_in[12];
  const float* l2_bhh = (const float*)d_in[13];
  const int*   ilen   = (const int*)d_in[14];
  float* out = (float*)d_out;

  char* ws = (char*)d_ws;
  size_t o = 0;
  auto carve = [&](size_t bytes) {
    char* p = ws + o;
    o = (o + bytes + 255) & ~(size_t)255;
    return p;
  };
  __bf16* encA    = (__bf16*)carve((size_t)B * S * H * 2);      // 8MB
  __bf16* encB    = (__bf16*)carve((size_t)B * S * H * 2);      // 8MB
  __bf16* wq      = (__bf16*)carve((size_t)H * H * 2);
  __bf16* wv      = (__bf16*)carve((size_t)H * H * 2);
  __bf16* w1ih    = (__bf16*)carve((size_t)H2 * G4 * 2);
  __bf16* w1hh    = (__bf16*)carve((size_t)H * G4 * 2);
  __bf16* w2ih    = (__bf16*)carve((size_t)H2 * G4 * 2);
  __bf16* w2hh    = (__bf16*)carve((size_t)H * G4 * 2);
  float*  pq      = (float*)carve((size_t)B * S * H * 4);       // 16MB
  float*  pv      = (float*)carve((size_t)B * S * H * 4);       // 16MB
  float*  scores  = (float*)carve((size_t)B * S * S * 4);       // 32MB
  __bf16* a_pack  = (__bf16*)carve((size_t)B * S * S * 2);      // 16MB
  __bf16* inp_all = (__bf16*)carve((size_t)S * B * H2 * 2);     // 16MB
  __bf16* x2ctx   = (__bf16*)carve((size_t)S * B * H * 2);      // 8MB

  // operand packing
  pack_enc_kernel<<<(B * S * H + 255) / 256, 256, 0, stream>>>(enc, encA);
  pack_encB_kernel<<<(B * S * H + 255) / 256, 256, 0, stream>>>(enc, encB);
  pack_attnW_kernel<<<(H * H + 255) / 256, 256, 0, stream>>>(attn_W, wq, wv);
  pack_weight_kernel<<<(G4 * H2 + 255) / 256, 256, 0, stream>>>(l1_Wih, w1ih,
                                                                G4, H2);
  pack_weight_kernel<<<(G4 * H + 255) / 256, 256, 0, stream>>>(l1_Whh, w1hh,
                                                               G4, H);
  pack_weight_kernel<<<(G4 * H2 + 255) / 256, 256, 0, stream>>>(l2_Wih, w2ih,
                                                                G4, H2);
  pack_weight_kernel<<<(G4 * H + 255) / 256, 256, 0, stream>>>(l2_Whh, w2hh,
                                                               G4, H);
  pack_inp_enc_kernel<<<(B * S * H + 255) / 256, 256, 0, stream>>>(enc,
                                                                   inp_all);

  // parallel attention pipeline
  gemm_pqpv_kernel<<<4096, 256, 0, stream>>>(encA, wq, wv, attn_b, pq, pv);
  score_kernel<<<dim3(S / 8, B), 256, 0, stream>>>(pq, pv, attn_V, ilen,
                                                   scores);
  softmax_pack_kernel<<<(B * S) / 32, 1024, 0, stream>>>(scores, a_pack);
  ctx_gemm_kernel<<<(B * (S / 16) * (H / 16)) / 8, 256, 0, stream>>>(
      a_pack, encB, inp_all, x2ctx);

  // serial recurrence (single WGP, WMMA gates, async-LDS operand streaming)
  decoder_kernel<<<1, 1024, 0, stream>>>(inp_all, x2ctx, w1ih, w1hh, w2ih,
                                         w2hh, l1_bih, l1_bhh, l2_bih, l2_bhh,
                                         ilen, bhs, bcs, out);
}
